// Module_59390807769684
// MI455X (gfx1250) — matrix operand order repaired and checked by reading, not compiled, not run on hardware
//
#include <hip/hip_runtime.h>

// ---- types -----------------------------------------------------------------
typedef __attribute__((ext_vector_type(16))) _Float16 v16h;
typedef __attribute__((ext_vector_type(8)))  _Float16 v8h;
typedef __attribute__((ext_vector_type(4)))  _Float16 v4h;
typedef __attribute__((ext_vector_type(8)))  float    v8f;
typedef __attribute__((ext_vector_type(4)))  float    v4f;

// ---- config ----------------------------------------------------------------
#define WIDTH     64
#define N_LAYERS  5          // in + 3 hidden + out (all 64x64)
#define WSTRIDE   72         // padded f16 row stride for LDS weights (16B mult)
#define ASTRIDE   72         // padded f16 row stride for LDS activations
#define WAVES     8
#define TILES     4          // 16-row tiles per wave
#define ROWS_PER_BLOCK (WAVES * TILES * 16)   // 512

__global__ __launch_bounds__(256)
void mlp_fused_wmma(const float* __restrict__ x,
                    const float* __restrict__ w_in,
                    const float* __restrict__ w_hidden,
                    const float* __restrict__ w_out,
                    float* __restrict__ out)
{
    // 45 KB weights + 72 KB activations = ~117 KB LDS (2 workgroups / WGP)
    __shared__ __align__(16) _Float16 sW[N_LAYERS * WIDTH * WSTRIDE];
    __shared__ __align__(16) _Float16 sAct[WAVES * TILES * 16 * ASTRIDE];

    const int tid  = threadIdx.x;
    const int wave = tid >> 5;
    const int lane = tid & 31;
    const int m16  = lane & 15;   // lane-within-half
    const int hs   = lane >> 4;   // half select

    // ---- cooperative weight load: W[o][i] (f32) -> W[o][i] (f16) in LDS ----
    for (int layer = 0; layer < N_LAYERS; ++layer) {
        const float* src = (layer == 0) ? w_in
                         : (layer == N_LAYERS - 1) ? w_out
                         : (w_hidden + (size_t)(layer - 1) * WIDTH * WIDTH);
        _Float16* dst = sW + layer * WIDTH * WSTRIDE;
        #pragma unroll
        for (int idx = tid; idx < WIDTH * WIDTH; idx += 256) {
            const int o = idx >> 6;
            const int i = idx & 63;
            dst[o * WSTRIDE + i] = (_Float16)src[idx];   // no transpose
        }
    }

    // ---- load x tiles once: f32 global (NT stream) -> f16 LDS ---------------
    const size_t rowBlock = (size_t)blockIdx.x * ROWS_PER_BLOCK;
    {
        const int col4  = lane & 15;   // which float4 in the row
        const int rpar  = lane >> 4;   // row parity within a row pair
        for (int t = 0; t < TILES; ++t) {
            const size_t rb = rowBlock + (size_t)(wave * TILES + t) * 16;
            _Float16* act = sAct + (wave * TILES + t) * 16 * ASTRIDE;
            #pragma unroll
            for (int r8 = 0; r8 < 8; ++r8) {
                const int r = r8 * 2 + rpar;
                v4f v = __builtin_nontemporal_load(
                            (const v4f*)(x + (rb + r) * WIDTH + col4 * 4));
                v4h h = __builtin_convertvector(v, v4h);
                *(v4h*)(act + r * ASTRIDE + col4 * 4) = h;  // ds_store_b64
            }
        }
    }
    __syncthreads();   // weights visible to all waves

    // ---- fused layer loop ---------------------------------------------------
    for (int layer = 0; layer < N_LAYERS; ++layer) {
        const _Float16* wt = sW + layer * WIDTH * WSTRIDE;

        // B fragments (32x16 f16, N = nt*16 + lane%16, K in elements), hoisted per layer:
        // lane<16 -> K octets {0-7,16-23}; lane>=16 -> {8-15,24-31}; kh adds 32 to K
        v16h B[4][2];
        #pragma unroll
        for (int nt = 0; nt < 4; ++nt) {
            #pragma unroll
            for (int kh = 0; kh < 2; ++kh) {
                const _Float16* p = wt + (nt * 16 + m16) * WSTRIDE + kh * 32 + hs * 8;
                v8h lo = *(const v8h*)p;          // ds_load_b128
                v8h hi = *(const v8h*)(p + 16);   // ds_load_b128
                B[nt][kh] = __builtin_shufflevector(lo, hi,
                    0,1,2,3,4,5,6,7,8,9,10,11,12,13,14,15);
            }
        }

        for (int t = 0; t < TILES; ++t) {
            _Float16* act = sAct + (wave * TILES + t) * 16 * ASTRIDE;

            // A fragments (16x32 f16): lane<16 -> M=lane, K octets {0-7,16-23};
            // lane>=16 -> K octets {8-15,24-31}; kh adds 32 to K.
            v16h A[2];
            #pragma unroll
            for (int kh = 0; kh < 2; ++kh) {
                const _Float16* p = act + m16 * ASTRIDE + kh * 32 + hs * 8;
                v8h lo = *(const v8h*)p;          // K base .. +7
                v8h hi = *(const v8h*)(p + 16);   // K base+16 .. +23
                A[kh] = __builtin_shufflevector(lo, hi,
                    0,1,2,3,4,5,6,7,8,9,10,11,12,13,14,15);
            }

            // 4 N-tiles x (2 accumulating WMMAs over K=64)
            v8f acc[4];
            #pragma unroll
            for (int nt = 0; nt < 4; ++nt) {
                v8f c = {};
                c = __builtin_amdgcn_wmma_f32_16x16x32_f16(
                        false, A[0], false, B[nt][0], (short)0, c, false, false);
                c = __builtin_amdgcn_wmma_f32_16x16x32_f16(
                        false, A[1], false, B[nt][1], (short)0, c, false, false);
                acc[nt] = c;
            }

            if (layer < N_LAYERS - 1) {
                // ReLU (single v_max_num_f32) + f16 cvt, back into same LDS tile.
                // C/D layout: lane<16 -> N=lane, M=v ; lane>=16 -> N=lane-16, M=v+8
                #pragma unroll
                for (int nt = 0; nt < 4; ++nt) {
                    #pragma unroll
                    for (int v = 0; v < 8; ++v) {
                        float f = __builtin_fmaxf(acc[nt][v], 0.0f);
                        act[(hs * 8 + v) * ASTRIDE + nt * 16 + m16] = (_Float16)f;
                    }
                }
            } else {
                // Linear output layer: NT-stream f32 result straight to global.
                const size_t rb = rowBlock + (size_t)(wave * TILES + t) * 16;
                #pragma unroll
                for (int nt = 0; nt < 4; ++nt) {
                    #pragma unroll
                    for (int v = 0; v < 8; ++v) {
                        __builtin_nontemporal_store(
                            acc[nt][v],
                            out + (rb + hs * 8 + v) * WIDTH + nt * 16 + m16);
                    }
                }
            }
        }
    }
}

extern "C" void kernel_launch(void* const* d_in, const int* in_sizes, int n_in,
                              void* d_out, int out_size, void* d_ws, size_t ws_size,
                              hipStream_t stream) {
    const float* x        = (const float*)d_in[0];
    const float* w_in     = (const float*)d_in[1];
    const float* w_hidden = (const float*)d_in[2];
    const float* w_out    = (const float*)d_in[3];
    float* out            = (float*)d_out;

    const int batch = in_sizes[0] / WIDTH;          // 1048576
    const int grid  = batch / ROWS_PER_BLOCK;       // 2048

    mlp_fused_wmma<<<grid, 256, 0, stream>>>(x, w_in, w_hidden, w_out, out);
}
